// PDFNetTruthChildren_75316546503337
// MI455X (gfx1250) — matrix operand order repaired and checked by reading, not compiled, not run on hardware
//
#include <hip/hip_runtime.h>

// ---------------------------------------------------------------------------
// MI455X / gfx1250 implementation.
//
// Input flattening (jax pytree, dict keys sorted, leaves b-then-w):
//   0:N_eta(8192) 1:N_energy 2:N_pT 3:N_phi 4:edge_index(2x262144 int32)
//   params (top-level keys sorted):
//   edge_mass: 5:l1.b(256) 6:l1.w(256x256) 7:l2.b(128) 8:l2.w(256x128)
//              9:l3.b(2) 10:l3.w(128x2)
//   en_dec:  11:l1.b 12:l1.w(1024x256) 13:l2.b(1) 14:l2.w(256)
//   en_enc:  15:l1.b 16:l1.w(1x256)   17:l2.b(1024) 18:l2.w(256x1024)
//   eta_dec: 19..22   eta_enc: 23..26
//   mass_mlp:27:l1.b 28:l1.w(5x256) 29:l2.b 30:l2.w(256x256) 31:l3.b 32:l3.w(256x256)
//   phi_dec: 33..36   phi_enc: 37..40   pt_dec: 41..44   pt_enc: 45..48
//
// Outputs flat: O_eta(8192) O_energy O_phi O_pT topo(270336x2) mass2(270336x2)
// ---------------------------------------------------------------------------

typedef _Float16 f16;
typedef __attribute__((ext_vector_type(16))) _Float16 v16h;
typedef __attribute__((ext_vector_type(8)))  _Float16 v8h;
typedef __attribute__((ext_vector_type(8)))  float    v8f;

#define N_NODES   8192
#define DEG       32
#define N_EDGES0  (N_NODES * DEG)      // 262144
#define N_EDGES   (N_EDGES0 + N_NODES) // 270336
#define M_TILE    128
#define THREADS   256                  // 8 wave32s
#define N_BLOCKS  (N_EDGES / M_TILE)   // 2112 exactly

// ---------------------------------------------------------------------------
// Kernel: per-node 4-momentum  (px,py,pz,E)
// ---------------------------------------------------------------------------
__global__ void pmu_dc_kernel(const float* eta, const float* en,
                              const float* pt, const float* phi, float* pmu) {
  int i = blockIdx.x * blockDim.x + threadIdx.x;
  if (i >= N_NODES) return;
  float p = pt[i], ph = phi[i];
  pmu[i * 4 + 0] = p * cosf(ph);
  pmu[i * 4 + 1] = p * sinf(ph);
  pmu[i * 4 + 2] = p * sinhf(eta[i]);
  pmu[i * 4 + 3] = en[i];
}

// ---------------------------------------------------------------------------
// Fold a scalar affine chain 1->256->1024->256->1 into (alpha, beta).
// One block (256 thr) per chain.
// ---------------------------------------------------------------------------
struct Chain { const float *w1, *b1, *w2, *b2, *w3, *b3, *w4, *b4; };
struct Chains4 { Chain c[4]; };

__global__ void fold_chain_kernel(Chains4 ch, float* alpha, float* beta) {
  __shared__ float a1[256], c1[256], a2[1024], c2[1024], a3[256], c3[256];
  const Chain C = ch.c[blockIdx.x];
  const int t = threadIdx.x;
  a1[t] = C.w1[t];
  c1[t] = C.b1[t];
  __syncthreads();
  for (int j = t; j < 1024; j += 256) {
    float sa = 0.f, sc = 0.f;
    for (int i = 0; i < 256; ++i) {
      float w = C.w2[i * 1024 + j];
      sa += a1[i] * w; sc += c1[i] * w;
    }
    a2[j] = sa; c2[j] = sc + C.b2[j];
  }
  __syncthreads();
  {
    float sa = 0.f, sc = 0.f;
    for (int i = 0; i < 1024; ++i) {
      float w = C.w3[i * 256 + t];
      sa += a2[i] * w; sc += c2[i] * w;
    }
    a3[t] = sa; c3[t] = sc + C.b3[t];
  }
  __syncthreads();
  a2[t] = a3[t] * C.w4[t];
  c2[t] = c3[t] * C.w4[t];
  __syncthreads();
  for (int s = 128; s > 0; s >>= 1) {
    if (t < s) { a2[t] += a2[t + s]; c2[t] += c2[t + s]; }
    __syncthreads();
  }
  if (t == 0) { alpha[blockIdx.x] = a2[0]; beta[blockIdx.x] = c2[0] + C.b4[0]; }
}

struct Apply4 { const float* x[4]; };
__global__ void apply_chain_kernel(Apply4 in, const float* alpha,
                                   const float* beta, float* out) {
  int i = blockIdx.x * blockDim.x + threadIdx.x;
  if (i >= 4 * N_NODES) return;
  int f = i >> 13, n = i & (N_NODES - 1);
  out[f * N_NODES + n] = alpha[f] * in.x[f][n] + beta[f];
}

// ---------------------------------------------------------------------------
// mass_mlp folding helpers (all affine): T = w2 @ w3 ; W_eff = w1 @ T ;
// b_eff = (b1 @ w2 + b2) @ w3 + b3
// ---------------------------------------------------------------------------
__global__ void mm256_kernel(const float* A, const float* B, float* T) {
  int idx = blockIdx.x * 256 + threadIdx.x;           // 65536 elems
  int r = idx >> 8, c = idx & 255;
  float s = 0.f;
  for (int k = 0; k < 256; ++k) s += A[r * 256 + k] * B[k * 256 + c];
  T[idx] = s;
}

__global__ void vecmat256_kernel(const float* v, const float* W,
                                 const float* badd, float* u) {
  int j = threadIdx.x;                                // 256 threads
  float s = badd[j];
  for (int i = 0; i < 256; ++i) s += v[i] * W[i * 256 + j];
  u[j] = s;
}

__global__ void weff_kernel(const float* w1, const float* T, float* weff) {
  int idx = blockIdx.x * 256 + threadIdx.x;           // 1280 elems
  if (idx >= 5 * 256) return;
  int f = idx >> 8, c = idx & 255;
  float s = 0.f;
  for (int k = 0; k < 256; ++k) s += w1[f * 256 + k] * T[k * 256 + c];
  weff[idx] = s;
}

// ---------------------------------------------------------------------------
// Weight swizzle: f32 row-major (KxN) -> f16 fragment-major.
// dst half index = ((kt*NT + nt)*32 + lane)*16 + q   with
//   r = q>>1, odd = q&1, hf = lane>>4, ml = lane&15
//   k = kt*32 + (q<8 ? hf*8 + q : 16 + hf*8 + (q-8)) ; col = nt*16 + ml
// So each lane's 16-half B fragment is one contiguous 32-byte chunk.
// ---------------------------------------------------------------------------
__global__ void swz_kernel(const float* W, f16* dst, int N, int NT, int total) {
  int idx = blockIdx.x * blockDim.x + threadIdx.x;
  if (idx >= total) return;
  int q    = idx & 15;
  int lane = (idx >> 4) & 31;
  int tile = idx >> 9;
  int nt = tile % NT;
  int kt = tile / NT;
  int r = q >> 1, odd = q & 1;
  int hf = lane >> 4, ml = lane & 15;
  int k   = kt * 32 + ((q < 8) ? (hf * 8 + q) : (16 + hf * 8 + (q - 8)));
  int col = nt * 16 + ml;
  dst[idx] = (f16)W[k * N + col];
}

// ---------------------------------------------------------------------------
// WMMA GEMM layer:  out(128xN,f16) = act(in(128xK,f16) @ W + bias)
// in  : LDS, row-major 128xK  -> A frag = two contiguous v8h (ds_load_b128 x2)
// Wswz: global, fragment-major -> B frag = one v16h (global_load_b128 x2)
// N processed in chunks of 8 column-tiles; per k-slab all 8 B fragments are
// loaded into distinct registers first (one load clause, one wait), then 8
// independent WMMAs issue back-to-back.
// ---------------------------------------------------------------------------
template <int K, int N, bool RELU>
__device__ inline void gemm_layer(const f16* __restrict__ in, f16* __restrict__ outh,
                                  const f16* __restrict__ Wswz,
                                  const float* __restrict__ bias,
                                  int wave, int lane) {
  constexpr int NT = N / 16;
  constexpr int KT = K / 32;
  constexpr int NC = (NT < 8) ? NT : 8;     // column tiles per chunk
  const int ml = lane & 15;
  const int hf = lane >> 4;
  const int m  = wave * 16 + ml;

  const v8f zero = {0.f, 0.f, 0.f, 0.f, 0.f, 0.f, 0.f, 0.f};

  for (int c0 = 0; c0 < NT; c0 += NC) {
    v8f acc[NC];
#pragma unroll
    for (int j = 0; j < NC; ++j) acc[j] = zero;

    for (int kt = 0; kt < KT; ++kt) {
      if (kt + 1 < KT)  // hint next K-slab of this chunk -> global_prefetch_b8
        __builtin_prefetch(Wswz + (((kt + 1) * NT + c0) * 32) * 16, 0, 3);

      // A fragment: row m, halves [kt*32+hf*8 ..+7] and [kt*32+16+hf*8 ..+7]
      const f16* arow = in + m * K + kt * 32 + hf * 8;
      v8h lo = *(const v8h*)(arow);
      v8h hi = *(const v8h*)(arow + 16);
      v16h a = __builtin_shufflevector(lo, hi, 0, 1, 2, 3, 4, 5, 6, 7,
                                       8, 9, 10, 11, 12, 13, 14, 15);

      // Load all NC B fragments of this slab into distinct registers.
      const f16* bbase = Wswz + ((kt * NT + c0) * 32 + lane) * 16;
      v16h b[NC];
#pragma unroll
      for (int j = 0; j < NC; ++j)
        b[j] = *(const v16h*)(bbase + j * 32 * 16);

      // Independent accumulators -> back-to-back WMMA burst.
#pragma unroll
      for (int j = 0; j < NC; ++j)
        acc[j] = __builtin_amdgcn_wmma_f32_16x16x32_f16(
            false, a, false, b[j], (short)0, acc[j], false, false);
    }

    // D (16x16 f32): VGPR r -> row hf*8+r, col ml
#pragma unroll
    for (int j = 0; j < NC; ++j) {
      int col = (c0 + j) * 16 + ml;
      float bv = bias[col];
#pragma unroll
      for (int r = 0; r < 8; ++r) {
        float v = acc[j][r] + bv;
        if (RELU) v = v > 0.f ? v : 0.f;
        outh[(wave * 16 + hf * 8 + r) * N + col] = (f16)v;
      }
    }
  }
}

// ---------------------------------------------------------------------------
// Fused per-edge MLP: X(5) -> [folded mass_mlp 5->256] -> relu(256->256)
//                     -> relu(256->128) -> (128->2)
// mode 0: build X from edges (Pmu[src]+Pmu[dst], mass);  mode 1: read X2.
// ---------------------------------------------------------------------------
__launch_bounds__(THREADS, 1)
__global__ void edge_mlp_kernel(int mode,
                                const int* __restrict__ edge_index,
                                const float* __restrict__ pmu,
                                const float* __restrict__ x2,
                                const float* __restrict__ w_eff,
                                const float* __restrict__ b_eff,
                                const f16* __restrict__ w1s, const float* __restrict__ b1,
                                const f16* __restrict__ w2s, const float* __restrict__ b2,
                                const float* __restrict__ w3, const float* __restrict__ b3,
                                float* __restrict__ out) {
  extern __shared__ char smem[];
  f16*   hA = (f16*)smem;                               // 128x256 (reused as 128x128)
  f16*   hB = (f16*)(smem + M_TILE * 256 * 2);          // 128x256
  float* xf = (float*)(smem + 2 * M_TILE * 256 * 2);    // 128x5
  float* wf = xf + M_TILE * 5;                          // 5x256 W_eff
  float* bf = wf + 5 * 256;                             // 256 b_eff

  const int tid  = threadIdx.x;
  const int wave = tid >> 5;
  const int lane = tid & 31;
  const int row0 = blockIdx.x * M_TILE;

  for (int i = tid; i < 5 * 256; i += THREADS) wf[i] = w_eff[i];
  for (int i = tid; i < 256; i += THREADS)     bf[i] = b_eff[i];

  if (mode == 0) {
    for (int r = tid; r < M_TILE; r += THREADS) {
      int e = row0 + r, s, d;
      if (e < N_EDGES0) { s = edge_index[e]; d = edge_index[N_EDGES0 + e]; }
      else              { s = e - N_EDGES0; d = s; }
      float px = pmu[s * 4 + 0] + pmu[d * 4 + 0];
      float py = pmu[s * 4 + 1] + pmu[d * 4 + 1];
      float pz = pmu[s * 4 + 2] + pmu[d * 4 + 2];
      float en = pmu[s * 4 + 3] + pmu[d * 4 + 3];
      float m2 = en * en - (px * px + py * py + pz * pz);
      float m  = m2 > 0.f ? sqrtf(m2) : 0.f;
      xf[r * 5 + 0] = px; xf[r * 5 + 1] = py; xf[r * 5 + 2] = pz;
      xf[r * 5 + 3] = en; xf[r * 5 + 4] = m;
    }
  } else {
    for (int i = tid; i < M_TILE * 5; i += THREADS) xf[i] = x2[row0 * 5 + i];
  }
  __syncthreads();

  // L0 (folded mass_mlp, K=5): plain FMA, output f16 into hA
  for (int idx = tid; idx < M_TILE * 256; idx += THREADS) {
    int r = idx >> 8, c = idx & 255;
    float s = bf[c];
#pragma unroll
    for (int f = 0; f < 5; ++f) s += xf[r * 5 + f] * wf[f * 256 + c];
    hA[idx] = (f16)s;
  }
  __syncthreads();

  gemm_layer<256, 256, true>(hA, hB, w1s, b1, wave, lane);   // L1
  __syncthreads();
  gemm_layer<256, 128, true>(hB, hA, w2s, b2, wave, lane);   // L2
  __syncthreads();

  // L3 (N=2): plain dot-128 per output
  for (int idx = tid; idx < M_TILE * 2; idx += THREADS) {
    int r = idx >> 1, c = idx & 1;
    float s = b3[c];
    for (int k = 0; k < 128; ++k) s += (float)hA[r * 128 + k] * w3[k * 2 + c];
    out[(row0 + r) * 2 + c] = s;
  }
}

#define SMEM_BYTES (2 * M_TILE * 256 * 2 + (M_TILE * 5 + 5 * 256 + 256) * 4)

// ---------------------------------------------------------------------------
// swi = argmax(topo, axis=1)  (strict >, ties -> 0)
// ---------------------------------------------------------------------------
__global__ void swi_kernel(const float* topo, int* swi) {
  int e = blockIdx.x * blockDim.x + threadIdx.x;
  if (e < N_EDGES) swi[e] = (topo[2 * e + 1] > topo[2 * e + 0]) ? 1 : 0;
}

// ---------------------------------------------------------------------------
// Per-node inclusive segment prefix (33 ordered entries: 32 edges + self loop)
// writes X2 rows at sorted position j = node*33 + t: [Psum(4), mass(Psum)]
// ---------------------------------------------------------------------------
__global__ void build_x2_kernel(const int* edge_index, const float* pmu,
                                const int* swi, float* x2) {
  int i = blockIdx.x * blockDim.x + threadIdx.x;
  if (i >= N_NODES) return;
  float ax = 0.f, ay = 0.f, az = 0.f, ae = 0.f;
  for (int t = 0; t < DEG + 1; ++t) {
    int e = (t < DEG) ? (i * DEG + t) : (N_EDGES0 + i);
    int d = (t < DEG) ? edge_index[N_EDGES0 + i * DEG + t] : i;
    if (swi[e]) {
      ax += pmu[d * 4 + 0]; ay += pmu[d * 4 + 1];
      az += pmu[d * 4 + 2]; ae += pmu[d * 4 + 3];
    }
    float m2 = ae * ae - (ax * ax + ay * ay + az * az);
    float m  = m2 > 0.f ? sqrtf(m2) : 0.f;
    long j = (long)i * (DEG + 1) + t;
    x2[j * 5 + 0] = ax; x2[j * 5 + 1] = ay; x2[j * 5 + 2] = az;
    x2[j * 5 + 3] = ae; x2[j * 5 + 4] = m;
  }
}

// ---------------------------------------------------------------------------
extern "C" void kernel_launch(void* const* d_in, const int* in_sizes, int n_in,
                              void* d_out, int out_size, void* d_ws, size_t ws_size,
                              hipStream_t stream) {
  const float* in_eta  = (const float*)d_in[0];
  const float* in_en   = (const float*)d_in[1];
  const float* in_pt   = (const float*)d_in[2];
  const float* in_phi  = (const float*)d_in[3];
  const int*   ei      = (const int*)d_in[4];
  float* out = (float*)d_out;

  // workspace layout (bytes)
  char* ws = (char*)d_ws;
  float* pmu   = (float*)(ws);                       // 8192x4        131072 B
  float* weff  = (float*)(ws + 131072);              // 5x256           5120 B
  float* beff  = (float*)(ws + 136192);              // 256             1024 B
  float* alpha = (float*)(ws + 137216);              // 4
  float* beta  = (float*)(ws + 137248);              // 4 (pad to 256)
  float* T     = (float*)(ws + 137472);              // 256x256       262144 B
  float* u     = (float*)(ws + 399616);              // 256             1024 B
  f16*   w1s   = (f16*)  (ws + 400640);              // 256x256 f16   131072 B (swizzled)
  f16*   w2s   = (f16*)  (ws + 531712);              // 256x128 f16    65536 B (swizzled)
  int*   swi   = (int*)  (ws + 597248);              // 270336 i32   1081344 B
  float* x2    = (float*)(ws + 1678592);             // 270336x5     5406720 B
  (void)in_sizes; (void)n_in; (void)out_size; (void)ws_size;

  // --- node 4-momenta --------------------------------------------------
  pmu_dc_kernel<<<N_NODES / 256, 256, 0, stream>>>(in_eta, in_en, in_pt, in_phi, pmu);

  // --- fold + apply the 4 pure-affine scalar chains ----------------------
  Chains4 ch;
  auto mk = [&](int enc0, int dec0) {
    Chain c;
    c.b1 = (const float*)d_in[enc0 + 0]; c.w1 = (const float*)d_in[enc0 + 1];
    c.b2 = (const float*)d_in[enc0 + 2]; c.w2 = (const float*)d_in[enc0 + 3];
    c.b3 = (const float*)d_in[dec0 + 0]; c.w3 = (const float*)d_in[dec0 + 1];
    c.b4 = (const float*)d_in[dec0 + 2]; c.w4 = (const float*)d_in[dec0 + 3];
    return c;
  };
  ch.c[0] = mk(23, 19);  // eta
  ch.c[1] = mk(15, 11);  // energy
  ch.c[2] = mk(37, 33);  // phi
  ch.c[3] = mk(45, 41);  // pt
  fold_chain_kernel<<<4, 256, 0, stream>>>(ch, alpha, beta);

  Apply4 ap;
  ap.x[0] = in_eta; ap.x[1] = in_en; ap.x[2] = in_phi; ap.x[3] = in_pt;
  apply_chain_kernel<<<(4 * N_NODES) / 256, 256, 0, stream>>>(ap, alpha, beta, out);

  // --- fold mass_mlp (pure affine 5->256) --------------------------------
  const float* m_b1 = (const float*)d_in[27];
  const float* m_w1 = (const float*)d_in[28];
  const float* m_b2 = (const float*)d_in[29];
  const float* m_w2 = (const float*)d_in[30];
  const float* m_b3 = (const float*)d_in[31];
  const float* m_w3 = (const float*)d_in[32];
  mm256_kernel<<<256, 256, 0, stream>>>(m_w2, m_w3, T);             // T = w2@w3
  vecmat256_kernel<<<1, 256, 0, stream>>>(m_b1, m_w2, m_b2, u);     // u = b1@w2+b2
  vecmat256_kernel<<<1, 256, 0, stream>>>(u, m_w3, m_b3, beff);     // beff = u@w3+b3
  weff_kernel<<<5, 256, 0, stream>>>(m_w1, T, weff);                // weff = w1@T

  // --- edge_mass weights -> swizzled fragment-major f16 -------------------
  const float* e_b1 = (const float*)d_in[5];
  const float* e_w1 = (const float*)d_in[6];
  const float* e_b2 = (const float*)d_in[7];
  const float* e_w2 = (const float*)d_in[8];
  const float* e_b3 = (const float*)d_in[9];
  const float* e_w3 = (const float*)d_in[10];
  swz_kernel<<<(256 * 256) / 256, 256, 0, stream>>>(e_w1, w1s, 256, 16, 256 * 256);
  swz_kernel<<<(256 * 128) / 256, 256, 0, stream>>>(e_w2, w2s, 128, 8, 256 * 128);

  // --- fused WMMA edge MLP, pass 1: topo ----------------------------------
  float* topo  = out + 4 * N_NODES;            // 270336 x 2
  float* mass2 = topo + 2 * N_EDGES;           // 270336 x 2
  hipFuncSetAttribute((const void*)edge_mlp_kernel,
                      hipFuncAttributeMaxDynamicSharedMemorySize, SMEM_BYTES);
  edge_mlp_kernel<<<N_BLOCKS, THREADS, SMEM_BYTES, stream>>>(
      0, ei, pmu, x2, weff, beff, w1s, e_b1, w2s, e_b2, e_w3, e_b3, topo);

  // --- switch + per-node segment prefix sums ------------------------------
  swi_kernel<<<(N_EDGES + 255) / 256, 256, 0, stream>>>(topo, swi);
  build_x2_kernel<<<N_NODES / 256, 256, 0, stream>>>(ei, pmu, swi, x2);

  // --- fused WMMA edge MLP, pass 2: mass2 ---------------------------------
  edge_mlp_kernel<<<N_BLOCKS, THREADS, SMEM_BYTES, stream>>>(
      1, ei, pmu, x2, weff, beff, w1s, e_b1, w2s, e_b2, e_w3, e_b3, mass2);
}
